// BeliefDecayDetector_81716047774242
// MI455X (gfx1250) — matrix operand order repaired and checked by reading, not compiled, not run on hardware
//
#include <hip/hip_runtime.h>
#include <math.h>

typedef __bf16 bf16;
typedef __attribute__((ext_vector_type(16))) __bf16 v16bf;
typedef __attribute__((ext_vector_type(8)))  __bf16 v8bf;
typedef __attribute__((ext_vector_type(8)))  float  v8f;

#define HIDN  384
#define NHEAD 4
#define DHEAD 96
#define NTOTN 8192
#define NEDGE 229376

__device__ __forceinline__ float gelu_f(float x){
  return 0.5f * x * (1.0f + erff(x * 0.7071067811865475f));
}
__device__ __forceinline__ float sigm_f(float x){ return 1.0f / (1.0f + __expf(-x)); }

__device__ __forceinline__ void atomicMaxF(float* addr, float val){
  int* ai = (int*)addr;
  int old = __float_as_int(*addr);
  while (val > __int_as_float(old)) {
    int assumed = old;
    old = atomicCAS(ai, assumed, __float_as_int(val));
    if (old == assumed) break;
  }
}

// ---------------------------------------------------------------------------
// Generic strided WMMA GEMM with LDS tile staging.
// C[m,n] = act( alpha*(*scale?)*(sum_k A[m,k]B[k,n]) + bias[n] )
// Block (256 thr = 8 waves) computes a 128x32 C macro-tile; each wave owns two
// 16x16 tiles (stacked in M) sharing one B fragment -> 2 WMMA per K-step.
// K stepped by 32 via v_wmma_f32_16x16x32_bf16; LDS tiles staged with b128
// fast paths when the contiguous stride is 1 (true for every GEMM here except
// the kvs A operand, which falls back to the strided scalar path).
// ---------------------------------------------------------------------------
__global__ void wmma_gemm_kernel(const float* __restrict__ A, int sAm, int sAk,
                                 const float* __restrict__ B, int sBk, int sBn,
                                 float* __restrict__ C, int sCm, int sCn,
                                 int M, int N, int K,
                                 const float* __restrict__ bias,
                                 const float* __restrict__ scalePtr,
                                 float alpha, int act)
{
  __shared__ __align__(16) bf16 sA[128 * 32];  // [m][k]
  __shared__ __align__(16) bf16 sB[32 * 32];   // [n][k] (transposed)

  const int tid  = threadIdx.x;
  const int wave = tid >> 5;
  const int lane = tid & 31;
  const int row  = lane & 15;
  const int half = lane >> 4;
  const int tilesN = (N + 31) >> 5;
  const int bm = blockIdx.x / tilesN;
  const int bn = blockIdx.x % tilesN;
  const int tm0 = bm << 7;            // 128-row macro tile
  const int tn0 = bn << 5;            // 32-col macro tile
  const int wmA = (wave >> 1) << 4;   // first wave subtile row (0..48)
  const int wmB = wmA + 64;           // second wave subtile row (64..112)
  const int wn  = (wave & 1) << 4;    // wave subtile col (0,16)

  const bool fastA  = (sAk == 1) && (tm0 + 128 <= M);
  const bool fastBn = (sBn == 1) && (tn0 + 32 <= N);
  const bool fastBk = (sBk == 1) && (tn0 + 32 <= N);

  v8f acc0 = {0.f,0.f,0.f,0.f,0.f,0.f,0.f,0.f};
  v8f acc1 = {0.f,0.f,0.f,0.f,0.f,0.f,0.f,0.f};

  for (int k0 = 0; k0 < K; k0 += 32) {
    // ---- stage A tile (128x32) ----
    if (fastA && (k0 + 32 <= K)) {
      const int m = tid >> 1;               // 0..127
      const int k = (tid & 1) << 4;         // 0,16
      const float4* src = (const float4*)(A + (size_t)(tm0 + m) * sAm + (k0 + k));
      float4 f0 = src[0], f1 = src[1], f2 = src[2], f3 = src[3];
      bf16* d = &sA[m * 32 + k];
      d[ 0] = (bf16)f0.x; d[ 1] = (bf16)f0.y; d[ 2] = (bf16)f0.z; d[ 3] = (bf16)f0.w;
      d[ 4] = (bf16)f1.x; d[ 5] = (bf16)f1.y; d[ 6] = (bf16)f1.z; d[ 7] = (bf16)f1.w;
      d[ 8] = (bf16)f2.x; d[ 9] = (bf16)f2.y; d[10] = (bf16)f2.z; d[11] = (bf16)f2.w;
      d[12] = (bf16)f3.x; d[13] = (bf16)f3.y; d[14] = (bf16)f3.z; d[15] = (bf16)f3.w;
      if (k0 + 64 <= K) __builtin_prefetch((const void*)(src + 8), 0, 3);
    } else {
      for (int i = tid; i < 128 * 32; i += 256) {
        int m = i >> 5, k = i & 31;
        int gm = tm0 + m, gk = k0 + k;
        float v = (gm < M && gk < K) ? A[(size_t)gm * sAm + (size_t)gk * sAk] : 0.f;
        sA[m * 32 + k] = (bf16)v;
      }
    }
    // ---- stage B tile (32k x 32n), transposed into sB[n][k] ----
    if (fastBn && (k0 + 32 <= K)) {
      const int k = tid >> 3;               // 0..31
      const int n = (tid & 7) << 2;         // 0..28
      const float4* src = (const float4*)(B + (size_t)(k0 + k) * sBk + (tn0 + n));
      float4 f = src[0];
      sB[(n + 0) * 32 + k] = (bf16)f.x;
      sB[(n + 1) * 32 + k] = (bf16)f.y;
      sB[(n + 2) * 32 + k] = (bf16)f.z;
      sB[(n + 3) * 32 + k] = (bf16)f.w;
    } else if (fastBk && (k0 + 32 <= K)) {
      const int n = tid >> 3;               // 0..31
      const int k = (tid & 7) << 2;         // 0..28
      const float4* src = (const float4*)(B + (size_t)(tn0 + n) * sBn + (k0 + k));
      float4 f = src[0];
      bf16* d = &sB[n * 32 + k];
      d[0] = (bf16)f.x; d[1] = (bf16)f.y; d[2] = (bf16)f.z; d[3] = (bf16)f.w;
    } else {
      for (int i = tid; i < 32 * 32; i += 256) {
        int n = i >> 5, k = i & 31;
        int gn = tn0 + n, gk = k0 + k;
        float v = (gn < N && gk < K) ? B[(size_t)gk * sBk + (size_t)gn * sBn] : 0.f;
        sB[n * 32 + k] = (bf16)v;
      }
    }
    __syncthreads();

    // ---- build fragments (ds_load_b128 pairs; B reused by both WMMAs) ----
    const v8bf* pb0 = (const v8bf*)(sB + (wn + row) * 32 + 8 * half);
    const v8bf* pb1 = (const v8bf*)(sB + (wn + row) * 32 + 16 + 8 * half);
    v8bf b0 = pb0[0], b1 = pb1[0];
    const v8bf* pa0 = (const v8bf*)(sA + (wmA + row) * 32 + 8 * half);
    const v8bf* pa1 = (const v8bf*)(sA + (wmA + row) * 32 + 16 + 8 * half);
    v8bf a0 = pa0[0], a1 = pa1[0];
    const v8bf* pc0 = (const v8bf*)(sA + (wmB + row) * 32 + 8 * half);
    const v8bf* pc1 = (const v8bf*)(sA + (wmB + row) * 32 + 16 + 8 * half);
    v8bf c0 = pc0[0], c1 = pc1[0];
    v16bf a, a2, b;
#pragma unroll
    for (int i = 0; i < 8; ++i) {
      a[i]  = a0[i]; a[i + 8]  = a1[i];
      a2[i] = c0[i]; a2[i + 8] = c1[i];
      b[i]  = b0[i]; b[i + 8]  = b1[i];
    }
    acc0 = __builtin_amdgcn_wmma_f32_16x16x32_bf16(false, a,  false, b, (short)0, acc0, false, false);
    acc1 = __builtin_amdgcn_wmma_f32_16x16x32_bf16(false, a2, false, b, (short)0, acc1, false, false);
    __syncthreads();
  }

  float sc = alpha;
  if (scalePtr) sc *= *scalePtr;
  const int n = tn0 + wn + row;
  float bv = (bias && n < N) ? bias[n] : 0.f;
#pragma unroll
  for (int r = 0; r < 8; ++r) {
    const int m0 = tm0 + wmA + r + 8 * half;
    if (m0 < M && n < N) {
      float v = acc0[r] * sc + bv;
      if (act == 1) v = gelu_f(v);
      else if (act == 2) v = fmaxf(v, 0.f);
      C[(size_t)m0 * sCm + (size_t)n * sCn] = v;
    }
    const int m1 = tm0 + wmB + r + 8 * half;
    if (m1 < M && n < N) {
      float v = acc1[r] * sc + bv;
      if (act == 1) v = gelu_f(v);
      else if (act == 2) v = fmaxf(v, 0.f);
      C[(size_t)m1 * sCm + (size_t)n * sCn] = v;
    }
  }
}

// ---------------------------------------------------------------------------
// Elementwise / utility kernels
// ---------------------------------------------------------------------------
__global__ void fill_kernel(float* p, size_t n, float v){
  size_t i = (size_t)blockIdx.x * blockDim.x + threadIdx.x;
  if (i < n) p[i] = v;
}
__global__ void copy_kernel(float* d, const float* s, size_t n){
  size_t i = (size_t)blockIdx.x * blockDim.x + threadIdx.x;
  if (i < n) d[i] = s[i];
}
__global__ void gelu_kernel(float* p, size_t n){
  size_t i = (size_t)blockIdx.x * blockDim.x + threadIdx.x;
  if (i < n) p[i] = gelu_f(p[i]);
}
__global__ void hgt_combine_kernel(float* dst, const float* o, const float* x,
                                   const float* skip, size_t n){
  size_t i = (size_t)blockIdx.x * blockDim.x + threadIdx.x;
  if (i >= n) return;
  float g = sigm_f(*skip);
  float y = g * o[i] + (1.f - g) * x[i];
  dst[i] = gelu_f(y);
}
__global__ void bnrelu_kernel(float* dst, const float* src, const float* g,
                              const float* b, size_t n, int cols){
  size_t i = (size_t)blockIdx.x * blockDim.x + threadIdx.x;
  if (i >= n) return;
  int c = (int)(i % cols);
  dst[i] = fmaxf(src[i] * (g[c] * 0.9999950000375f) + b[c], 0.f);
}
__global__ void mix_kernel(float* dst, const float* a, const float* b2,
                           float ca, float cb, size_t n){
  size_t i = (size_t)blockIdx.x * blockDim.x + threadIdx.x;
  if (i < n) dst[i] = ca * a[i] + cb * b2[i];
}
__global__ void concat2_rows_kernel(float* dst, const float* a, const float* b2,
                                    int rows, int ca, int cb){
  size_t i = (size_t)blockIdx.x * blockDim.x + threadIdx.x;
  size_t tot = (size_t)rows * (ca + cb);
  if (i >= tot) return;
  int w = ca + cb;
  int r = (int)(i / w), c = (int)(i % w);
  dst[i] = (c < ca) ? a[(size_t)r * ca + c] : b2[(size_t)r * cb + (c - ca)];
}
__global__ void build_dg_kernel(int* dg, const int* edst, int cnt, int offD, int eoff){
  int e = blockIdx.x * blockDim.x + threadIdx.x;
  if (e < cnt) dg[eoff + e] = offD + edst[eoff + e];
}

// ---------------------------------------------------------------------------
// HGT edge phase
// ---------------------------------------------------------------------------
__global__ void hgt_edge_attn_kernel(const float* __restrict__ KQV,
                                     const int* __restrict__ esrc,
                                     const int* __restrict__ edst,
                                     const float* __restrict__ krel,   // (H,96,96) base for (l,r)
                                     const float* __restrict__ prel,   // (H,)
                                     float* __restrict__ abuf,
                                     float* __restrict__ mx,
                                     int cnt, int offS, int offD, int eoff)
{
  __shared__ float srel[DHEAD * DHEAD];
  const int h = blockIdx.y;
  const float* rel = krel + (size_t)h * DHEAD * DHEAD;
  for (int i = threadIdx.x; i < DHEAD * DHEAD; i += blockDim.x) srel[i] = rel[i];
  __syncthreads();
  int e = blockIdx.x * blockDim.x + threadIdx.x;
  if (e >= cnt) return;
  int eg = eoff + e;
  int si = esrc[eg], di = edst[eg];
  const float* krow = KQV + (size_t)(offS + si) * (3 * HIDN) + h * DHEAD;
  const float* qrow = KQV + (size_t)(offD + di) * (3 * HIDN) + HIDN + h * DHEAD;
  float acc = 0.f;
  for (int f = 0; f < DHEAD; ++f) {
    float ke = 0.f;
    for (int d = 0; d < DHEAD; ++d) ke = fmaf(krow[d], srel[d * DHEAD + f], ke);
    acc = fmaf(qrow[f], ke, acc);
  }
  float a = acc * prel[h] * 0.10206207261596575f;  // 1/sqrt(96)
  abuf[(size_t)eg * NHEAD + h] = a;
  atomicMaxF(&mx[(size_t)(offD + di) * NHEAD + h], a);
}

__global__ void edge_expden_kernel(float* __restrict__ abuf, const float* __restrict__ mx,
                                   float* __restrict__ den, const int* __restrict__ dg)
{
  size_t idx = (size_t)blockIdx.x * blockDim.x + threadIdx.x;
  if (idx >= (size_t)NEDGE * NHEAD) return;
  size_t e = idx >> 2; int h = (int)(idx & 3);
  int d = dg[e];
  float ex = __expf(abuf[idx] - mx[(size_t)d * NHEAD + h]);
  abuf[idx] = ex;
  atomicAdd(&den[(size_t)d * NHEAD + h], ex);
}

__global__ void hgt_edge_agg_kernel(const float* __restrict__ KQV,
                                    const int* __restrict__ esrc,
                                    const int* __restrict__ dg,
                                    const float* __restrict__ vrel,
                                    const float* __restrict__ abuf,
                                    const float* __restrict__ den,
                                    float* __restrict__ agg,
                                    int cnt, int offS, int eoff)
{
  __shared__ float srel[DHEAD * DHEAD];
  const int h = blockIdx.y;
  const float* rel = vrel + (size_t)h * DHEAD * DHEAD;
  for (int i = threadIdx.x; i < DHEAD * DHEAD; i += blockDim.x) srel[i] = rel[i];
  __syncthreads();
  int e = blockIdx.x * blockDim.x + threadIdx.x;
  if (e >= cnt) return;
  int eg = eoff + e;
  int si = esrc[eg];
  int d  = dg[eg];
  const float* vrow = KQV + (size_t)(offS + si) * (3 * HIDN) + 2 * HIDN + h * DHEAD;
  float w = abuf[(size_t)eg * NHEAD + h] / den[(size_t)d * NHEAD + h];
  float* out = agg + (size_t)d * HIDN + h * DHEAD;
  for (int f = 0; f < DHEAD; ++f) {
    float ve = 0.f;
    for (int dd = 0; dd < DHEAD; ++dd) ve = fmaf(vrow[dd], srel[dd * DHEAD + f], ve);
    atomicAdd(&out[f], w * ve);
  }
}

// ---------------------------------------------------------------------------
// Row reductions: LayerNorm and softmax
// ---------------------------------------------------------------------------
__global__ void ln_rows_kernel(float* dst, const float* src, const float* g,
                               const float* b, int cols, int gelu_after)
{
  __shared__ float red[128];
  const int t = threadIdx.x;
  const float* x = src + (size_t)blockIdx.x * cols;
  float s = 0.f;
  for (int c = t; c < cols; c += 128) s += x[c];
  red[t] = s; __syncthreads();
  for (int st = 64; st > 0; st >>= 1){ if (t < st) red[t] += red[t + st]; __syncthreads(); }
  float mean = red[0] / cols; __syncthreads();
  float v = 0.f;
  for (int c = t; c < cols; c += 128){ float d0 = x[c] - mean; v += d0 * d0; }
  red[t] = v; __syncthreads();
  for (int st = 64; st > 0; st >>= 1){ if (t < st) red[t] += red[t + st]; __syncthreads(); }
  float rs = rsqrtf(red[0] / cols + 1e-5f);
  float* y = dst + (size_t)blockIdx.x * cols;
  for (int c = t; c < cols; c += 128){
    float val = (x[c] - mean) * rs * g[c] + b[c];
    if (gelu_after) val = gelu_f(val);
    y[c] = val;
  }
}

__global__ void softmax_rows_kernel(float* s, int cols)
{
  __shared__ float red[256];
  const int t = threadIdx.x;
  float* p = s + (size_t)blockIdx.x * cols;
  float mx = -1e30f;
  for (int c = t; c < cols; c += 256) mx = fmaxf(mx, p[c]);
  red[t] = mx; __syncthreads();
  for (int st = 128; st > 0; st >>= 1){ if (t < st) red[t] = fmaxf(red[t], red[t + st]); __syncthreads(); }
  float m = red[0]; __syncthreads();
  float sum = 0.f;
  for (int c = t; c < cols; c += 256){ float e = __expf(p[c] - m); p[c] = e; sum += e; }
  red[t] = sum; __syncthreads();
  for (int st = 128; st > 0; st >>= 1){ if (t < st) red[t] += red[t + st]; __syncthreads(); }
  float inv = 1.f / red[0];
  for (int c = t; c < cols; c += 256) p[c] *= inv;
}

// ---------------------------------------------------------------------------
// Linear-transformer helpers
// ---------------------------------------------------------------------------
__global__ void sumsq_kernel(const float* x, size_t n, float* out){
  __shared__ float red[256];
  const int t = threadIdx.x;
  float s = 0.f;
  for (size_t i = (size_t)blockIdx.x * 256 + t; i < n; i += (size_t)gridDim.x * 256){
    float v = x[i]; s += v * v;
  }
  red[t] = s; __syncthreads();
  for (int st = 128; st > 0; st >>= 1){ if (t < st) red[t] += red[t + st]; __syncthreads(); }
  if (t == 0) atomicAdd(out, red[0]);
}
__global__ void invnorm_kernel(float* scal){
  if (threadIdx.x == 0)
    scal[2] = rsqrtf(fmaxf(scal[0], 1e-30f)) * rsqrtf(fmaxf(scal[1], 1e-30f));
}
__global__ void colsum_kernel(const float* x, float* out, int rows, int cols){
  int c = blockIdx.x * blockDim.x + threadIdx.x;
  if (c >= cols) return;
  float s = 0.f;
  for (int r = 0; r < rows; ++r) s += x[(size_t)r * cols + c];
  out[c] = s;
}
__global__ void dnm_kernel(const float* qb, const float* ksum, const float* scal,
                           float* dnm, int Ns){
  int idx = blockIdx.x * blockDim.x + threadIdx.x;
  if (idx >= Ns * NHEAD) return;
  int n = idx >> 2, h = idx & 3;
  const float* q = qb + (size_t)n * (NHEAD * HIDN) + h * HIDN;
  const float* ks = ksum + h * HIDN;
  float s = 0.f;
  for (int m = 0; m < HIDN; ++m) s = fmaf(q[m], ks[m], s);
  dnm[idx] = s * (*scal) + (float)Ns;
}
__global__ void linattn_update_kernel(float* xg, const float* numb, const float* vb,
                                      const float* dnm, const float* g, const float* b, int Ns){
  size_t idx = (size_t)blockIdx.x * blockDim.x + threadIdx.x;
  if (idx >= (size_t)Ns * HIDN) return;
  int n = (int)(idx / HIDN), m = (int)(idx % HIDN);
  float z = 0.f;
  for (int h = 0; h < NHEAD; ++h){
    size_t o = (size_t)n * (NHEAD * HIDN) + h * HIDN + m;
    z += (numb[o] + (float)Ns * vb[o]) / dnm[n * NHEAD + h];
  }
  z *= 0.25f;
  float xv = 0.5f * z + 0.5f * xg[idx];
  xg[idx] = fmaxf(xv * (g[m] * 0.9999950000375f) + b[m], 0.f);
}

// ---------------------------------------------------------------------------
// Conv1d(k=3, pad=1) + GELU; means; GEMVs
// ---------------------------------------------------------------------------
__global__ void conv1d_k3_gelu_kernel(const float* in, int sIC, int sP,
                                      const float* W, const float* bias,
                                      float* out, int IC, int OC, int L){
  size_t idx = (size_t)blockIdx.x * blockDim.x + threadIdx.x;
  if (idx >= (size_t)OC * L) return;
  int oc = (int)(idx / L), p = (int)(idx % L);
  float s = bias[oc];
  for (int ic = 0; ic < IC; ++ic){
    const float* w = W + ((size_t)oc * IC + ic) * 3;
    const float* ip = in + (size_t)ic * sIC;
#pragma unroll
    for (int t = 0; t < 3; ++t){
      int pp = p + t - 1;
      if (pp >= 0 && pp < L) s = fmaf(ip[(size_t)pp * sP], w[t], s);
    }
  }
  out[(size_t)oc * L + p] = gelu_f(s);
}
__global__ void col_mean_kernel(const float* x, int sr, int sc, int rows, int cols, float* out){
  int c = blockIdx.x * blockDim.x + threadIdx.x;
  if (c >= cols) return;
  float s = 0.f;
  for (int r = 0; r < rows; ++r) s += x[(size_t)r * sr + (size_t)c * sc];
  out[c] = s / rows;
}
__global__ void vecmat2_kernel(const float* x1, int n1, const float* x2, int n2,
                               const float* W, int ldw, const float* bias,
                               float* out, int N, int act){
  int n = blockIdx.x * blockDim.x + threadIdx.x;
  if (n >= N) return;
  float s = bias ? bias[n] : 0.f;
  for (int k = 0; k < n1; ++k) s = fmaf(x1[k], W[(size_t)k * ldw + n], s);
  for (int k = 0; k < n2; ++k) s = fmaf(x2[k], W[(size_t)(n1 + k) * ldw + n], s);
  if (act == 2) s = fmaxf(s, 0.f);
  out[n] = s;
}
__global__ void gemv_rows_kernel(const float* X, int ldx, const float* W, int ldw,
                                 const float* bias, float* out, int rows, int K, int act){
  int r = blockIdx.x * blockDim.x + threadIdx.x;
  if (r >= rows) return;
  float s = bias[0];
  const float* x = X + (size_t)r * ldx;
  for (int k = 0; k < K; ++k) s = fmaf(x[k], W[(size_t)k * ldw], s);
  if (act == 3) s = sigm_f(s);
  out[r] = s;
}

// ---------------------------------------------------------------------------
// Final classifier head (single block)
// ---------------------------------------------------------------------------
__global__ void final_head_kernel(const float* te, const float* ap, const float* bc,
                                  const float* sd,
                                  const float* W1, const float* b1,
                                  const float* lng, const float* lnb,
                                  const float* W2, const float* b2,
                                  const float* W3, const float* b3, float* out)
{
  __shared__ float comb[1344];
  __shared__ float z1[384];
  __shared__ float z2[192];
  __shared__ float red[512];
  const int t = threadIdx.x;
  if (t < 384){ comb[t] = te[t]; comb[384 + t] = ap[t]; comb[768 + t] = bc[t]; }
  if (t < 192) comb[1152 + t] = sd[t];
  __syncthreads();
  float v1 = 0.f;
  if (t < 384){
    float s = b1[t];
    for (int k = 0; k < 1344; ++k) s = fmaf(comb[k], W1[(size_t)k * 384 + t], s);
    v1 = s;
  }
  red[t] = (t < 384) ? v1 : 0.f; __syncthreads();
  for (int st = 256; st > 0; st >>= 1){ if (t < st) red[t] += red[t + st]; __syncthreads(); }
  float mean = red[0] / 384.f; __syncthreads();
  float dv = (t < 384) ? (v1 - mean) : 0.f;
  red[t] = dv * dv; __syncthreads();
  for (int st = 256; st > 0; st >>= 1){ if (t < st) red[t] += red[t + st]; __syncthreads(); }
  float rs = rsqrtf(red[0] / 384.f + 1e-5f); __syncthreads();
  if (t < 384) z1[t] = fmaxf((v1 - mean) * rs * lng[t] + lnb[t], 0.f);
  __syncthreads();
  if (t < 192){
    float s = b2[t];
    for (int k = 0; k < 384; ++k) s = fmaf(z1[k], W2[(size_t)k * 192 + t], s);
    z2[t] = fmaxf(s, 0.f);
  }
  __syncthreads();
  red[t] = (t < 192) ? z2[t] * W3[t] : 0.f; __syncthreads();
  for (int st = 256; st > 0; st >>= 1){ if (t < st) red[t] += red[t + st]; __syncthreads(); }
  if (t == 0){ float lg = red[0] + b3[0]; out[0] = lg; out[1] = sigm_f(lg); }
}

// ---------------------------------------------------------------------------
// Host helpers
// ---------------------------------------------------------------------------
static inline dim3 g1(size_t n){ return dim3((unsigned)((n + 255) / 256)); }

static void launch_gemm(hipStream_t st, const float* A, int sAm, int sAk,
                        const float* B, int sBk, int sBn,
                        float* C, int sCm, int sCn,
                        int M, int N, int K,
                        const float* bias, const float* scalePtr,
                        float alpha, int act){
  unsigned blocks = (unsigned)(((M + 127) / 128) * (long)((N + 31) / 32));
  wmma_gemm_kernel<<<dim3(blocks), dim3(256), 0, st>>>(A, sAm, sAk, B, sBk, sBn,
      C, sCm, sCn, M, N, K, bias, scalePtr, alpha, act);
}

static void run_mha(hipStream_t st, const float* qin, int Nq, const float* kvin, int Nk,
                    const float* Wqkv, const float* bqkv, const float* Wo, const float* bo,
                    float* qb, float* kb, float* vb, float* ob, float* scores, float* outp)
{
  launch_gemm(st, qin, 384, 1, Wqkv,       1152, 1, qb, 384, 1, Nq, 384, 384, bqkv,       nullptr, 1.f, 0);
  launch_gemm(st, kvin, 384, 1, Wqkv + 384, 1152, 1, kb, 384, 1, Nk, 384, 384, bqkv + 384, nullptr, 1.f, 0);
  launch_gemm(st, kvin, 384, 1, Wqkv + 768, 1152, 1, vb, 384, 1, Nk, 384, 384, bqkv + 768, nullptr, 1.f, 0);
  for (int h = 0; h < NHEAD; ++h)
    launch_gemm(st, qb + h * 96, 384, 1, kb + h * 96, 1, 384,
                scores + (size_t)h * Nq * Nk, Nk, 1, Nq, Nk, 96,
                nullptr, nullptr, 0.10206207261596575f, 0);
  softmax_rows_kernel<<<dim3(NHEAD * Nq), dim3(256), 0, st>>>(scores, Nk);
  for (int h = 0; h < NHEAD; ++h)
    launch_gemm(st, scores + (size_t)h * Nq * Nk, Nk, 1, vb + h * 96, 384, 1,
                ob + h * 96, 384, 1, Nq, 96, Nk, nullptr, nullptr, 1.f, 0);
  launch_gemm(st, ob, 384, 1, Wo, 384, 1, outp, 384, 1, Nq, 384, 384, bo, nullptr, 1.f, 0);
}

// ---------------------------------------------------------------------------
extern "C" void kernel_launch(void* const* d_in, const int* in_sizes, int n_in,
                              void* d_out, int out_size, void* d_ws, size_t ws_size,
                              hipStream_t stream)
{
  (void)in_sizes; (void)n_in; (void)out_size; (void)ws_size;
  const float* xin[5]  = {(const float*)d_in[0], (const float*)d_in[1], (const float*)d_in[2],
                          (const float*)d_in[3], (const float*)d_in[4]};
  const float* hgt_Wkqv = (const float*)d_in[5];
  const float* hgt_bkqv = (const float*)d_in[6];
  const float* hgt_Wout = (const float*)d_in[7];
  const float* hgt_bout = (const float*)d_in[8];
  const float* hgt_krel = (const float*)d_in[9];
  const float* hgt_vrel = (const float*)d_in[10];
  const float* hgt_prel = (const float*)d_in[11];
  const float* hgt_skip = (const float*)d_in[12];
  const float* ln_g = (const float*)d_in[13];
  const float* ln_b = (const float*)d_in[14];
  const float* sp_W = (const float*)d_in[15];
  const float* sp_b = (const float*)d_in[16];
  const float* sp_lng = (const float*)d_in[17];
  const float* sp_lnb = (const float*)d_in[18];
  const float* fc0_W = (const float*)d_in[19];
  const float* fc0_b = (const float*)d_in[20];
  const float* bn_g = (const float*)d_in[21];
  const float* bn_b = (const float*)d_in[22];
  const float* tq_W = (const float*)d_in[23];
  const float* tq_b = (const float*)d_in[24];
  const float* tk_W = (const float*)d_in[25];
  const float* tk_b = (const float*)d_in[26];
  const float* tv_W = (const float*)d_in[27];
  const float* tv_b = (const float*)d_in[28];
  const float* mha_Wqkv = (const float*)d_in[29];
  const float* mha_bqkv = (const float*)d_in[30];
  const float* mha_Wo = (const float*)d_in[31];
  const float* mha_bo = (const float*)d_in[32];
  const float* c1_W = (const float*)d_in[33];
  const float* c1_b = (const float*)d_in[34];
  const float* c2_W = (const float*)d_in[35];
  const float* c2_b = (const float*)d_in[36];
  const float* op_W = (const float*)d_in[37];
  const float* op_b = (const float*)d_in[38];
  const float* as1_W = (const float*)d_in[39];
  const float* as1_b = (const float*)d_in[40];
  const float* as2_W = (const float*)d_in[41];
  const float* as2_b = (const float*)d_in[42];
  const float* sc1_W = (const float*)d_in[43];
  const float* sc1_b = (const float*)d_in[44];
  const float* sc2_W = (const float*)d_in[45];
  const float* sc2_b = (const float*)d_in[46];
  const float* cl1_W = (const float*)d_in[47];
  const float* cl1_b = (const float*)d_in[48];
  const float* cl_lng = (const float*)d_in[49];
  const float* cl_lnb = (const float*)d_in[50];
  const float* cl2_W = (const float*)d_in[51];
  const float* cl2_b = (const float*)d_in[52];
  const float* cl3_W = (const float*)d_in[53];
  const float* cl3_b = (const float*)d_in[54];
  const float* tp_W = (const float*)d_in[55];
  const float* tp_b = (const float*)d_in[56];
  const int* edge_src = (const int*)d_in[57];
  const int* edge_dst = (const int*)d_in[58];
  float* out = (float*)d_out;
  float* ws = (float*)d_ws;
  hipStream_t st = stream;

  static const int OFF[5] = {0, 2048, 4096, 6144, 7168};
  static const int NT[5]  = {2048, 2048, 2048, 1024, 1024};
  static const int ES[8]  = {0, 1, 1, 2, 0, 3, 1, 4};
  static const int ED[8]  = {1, 0, 2, 2, 3, 1, 4, 2};
  static const int EC[8]  = {32768, 32768, 32768, 32768, 16384, 32768, 16384, 32768};
  static const int EO[8]  = {0, 32768, 65536, 98304, 131072, 147456, 180224, 196608};

  // ---- workspace layout (floats) ----
  size_t cur = 0;
  auto take = [&](size_t n){ size_t r = cur; cur += n; return r; };
  const size_t XS_CUR_ = take(8192ull * 384);
  const size_t XS_NEW_ = take(8192ull * 384);
  const size_t KQV_    = take(8192ull * 1152);
  const size_t ABUF_   = take((size_t)NEDGE * 4);
  const size_t DGBUF_  = take((size_t)NEDGE);
  const size_t MX_     = take(8192ull * 4);
  const size_t DEN_    = take(8192ull * 4);
  const size_t AGG_    = take(8192ull * 384);
  const size_t HS_     = take(2048ull * 384);
  const size_t XG_     = take(2048ull * 384);
  const size_t QB_     = take(2048ull * 1536);
  const size_t KB_     = take(2048ull * 1536);
  const size_t VB_     = take(2048ull * 1536);
  const size_t NUMB_   = take(2048ull * 1536);
  const size_t KVS_    = take(4ull * 384 * 384);
  const size_t KSUM_   = take(1536);
  const size_t DNM_    = take(2048ull * 4);
  const size_t SCAL_   = take(16);
  const size_t HMIX_   = take(2048ull * 384);
  const size_t SCORES_ = take(4ull * 2048 * 2048);
  const size_t QB2_    = take(2048ull * 384);
  const size_t KB2_    = take(2048ull * 384);
  const size_t VB2_    = take(2048ull * 384);
  const size_t OB2_    = take(2048ull * 384);
  const size_t TRAJ_   = take(2048ull * 384);
  const size_t AICTX_  = take(2048ull * 384);
  const size_t BCTX_   = take(2048ull * 384);
  const size_t CONV1_  = take(384ull * 2048);
  const size_t CONV2_  = take(192ull * 2048);
  const size_t SMALL_  = take(4096);
  const size_t COMB_   = take(2048ull * 768);
  const size_t ASBUF_  = take(2048ull * 384);
  int* dg = (int*)(ws + DGBUF_);

  // ---- init node features and dg ----
  for (int t = 0; t < 5; ++t)
    copy_kernel<<<g1((size_t)NT[t] * 384), 256, 0, st>>>(ws + XS_CUR_ + (size_t)OFF[t] * 384,
                                                         xin[t], (size_t)NT[t] * 384);
  for (int r = 0; r < 8; ++r)
    build_dg_kernel<<<g1(EC[r]), 256, 0, st>>>(dg, edge_dst, EC[r], OFF[ED[r]], EO[r]);

  // ---- HGT layers ----
  for (int l = 0; l < 2; ++l){
    for (int t = 0; t < 5; ++t)
      launch_gemm(st, ws + XS_CUR_ + (size_t)OFF[t] * 384, 384, 1,
                  hgt_Wkqv + (size_t)(l * 5 + t) * 384 * 1152, 1152, 1,
                  ws + KQV_ + (size_t)OFF[t] * 1152, 1152, 1,
                  NT[t], 1152, 384, hgt_bkqv + (size_t)(l * 5 + t) * 1152, nullptr, 1.f, 0);
    fill_kernel<<<g1(8192ull * 4), 256, 0, st>>>(ws + MX_, 8192ull * 4, -1e30f);
    fill_kernel<<<g1(8192ull * 4), 256, 0, st>>>(ws + DEN_, 8192ull * 4, 0.f);
    fill_kernel<<<g1(8192ull * 384), 256, 0, st>>>(ws + AGG_, 8192ull * 384, 0.f);
    for (int r = 0; r < 8; ++r)
      hgt_edge_attn_kernel<<<dim3((EC[r] + 127) / 128, 4), 128, 0, st>>>(
          ws + KQV_, edge_src, edge_dst,
          hgt_krel + (size_t)(l * 8 + r) * 4 * 9216, hgt_prel + (size_t)(l * 8 + r) * 4,
          ws + ABUF_, ws + MX_, EC[r], OFF[ES[r]], OFF[ED[r]], EO[r]);
    edge_expden_kernel<<<g1((size_t)NEDGE * 4), 256, 0, st>>>(ws + ABUF_, ws + MX_, ws + DEN_, dg);
    for (int r = 0; r < 8; ++r)
      hgt_edge_agg_kernel<<<dim3((EC[r] + 127) / 128, 4), 128, 0, st>>>(
          ws + KQV_, edge_src, dg, hgt_vrel + (size_t)(l * 8 + r) * 4 * 9216,
          ws + ABUF_, ws + DEN_, ws + AGG_, EC[r], OFF[ES[r]], EO[r]);
    gelu_kernel<<<g1(8192ull * 384), 256, 0, st>>>(ws + AGG_, 8192ull * 384);
    for (int t = 0; t < 5; ++t)
      launch_gemm(st, ws + AGG_ + (size_t)OFF[t] * 384, 384, 1,
                  hgt_Wout + (size_t)(l * 5 + t) * 384 * 384, 384, 1,
                  ws + XS_NEW_ + (size_t)OFF[t] * 384, 384, 1,
                  NT[t], 384, 384, hgt_bout + (size_t)(l * 5 + t) * 384, nullptr, 1.f, 0);
    for (int t = 0; t < 5; ++t)
      hgt_combine_kernel<<<g1((size_t)NT[t] * 384), 256, 0, st>>>(
          ws + XS_NEW_ + (size_t)OFF[t] * 384, ws + XS_NEW_ + (size_t)OFF[t] * 384,
          ws + XS_CUR_ + (size_t)OFF[t] * 384, hgt_skip + l * 5 + t, (size_t)NT[t] * 384);
    for (int t = 0; t < 5; ++t)
      ln_rows_kernel<<<dim3(NT[t]), 128, 0, st>>>(
          ws + XS_CUR_ + (size_t)OFF[t] * 384, ws + XS_NEW_ + (size_t)OFF[t] * 384,
          ln_g + t * 384, ln_b + t * 384, 384, 0);
  }
  const float* user_h   = ws + XS_CUR_;
  const float* ai_h     = ws + XS_CUR_ + 2048ull * 384;
  const float* stance_h = ws + XS_CUR_ + 4096ull * 384;
  const float* belief_h = ws + XS_CUR_ + 7168ull * 384;

  // ---- stance path: hs, xg ----
  launch_gemm(st, stance_h, 384, 1, sp_W, 384, 1, ws + XS_NEW_, 384, 1,
              2048, 384, 384, sp_b, nullptr, 1.f, 0);
  ln_rows_kernel<<<dim3(2048), 128, 0, st>>>(ws + HS_, ws + XS_NEW_, sp_lng, sp_lnb, 384, 1);
  launch_gemm(st, ws + HS_, 384, 1, fc0_W, 384, 1, ws + XS_NEW_, 384, 1,
              2048, 384, 384, fc0_b, nullptr, 1.f, 0);
  bnrelu_kernel<<<g1(2048ull * 384), 256, 0, st>>>(ws + XG_, ws + XS_NEW_, bn_g, bn_b,
                                                   2048ull * 384, 384);
  // ---- linear-transformer blocks ----
  for (int i = 0; i < 2; ++i){
    launch_gemm(st, ws + XG_, 384, 1, tq_W + (size_t)i * 384 * 1536, 1536, 1,
                ws + QB_, 1536, 1, 2048, 1536, 384, tq_b + (size_t)i * 1536, nullptr, 1.f, 0);
    launch_gemm(st, ws + XG_, 384, 1, tk_W + (size_t)i * 384 * 1536, 1536, 1,
                ws + KB_, 1536, 1, 2048, 1536, 384, tk_b + (size_t)i * 1536, nullptr, 1.f, 0);
    launch_gemm(st, ws + XG_, 384, 1, tv_W + (size_t)i * 384 * 1536, 1536, 1,
                ws + VB_, 1536, 1, 2048, 1536, 384, tv_b + (size_t)i * 1536, nullptr, 1.f, 0);
    fill_kernel<<<dim3(1), 256, 0, st>>>(ws + SCAL_, 4, 0.f);
    sumsq_kernel<<<dim3(512), 256, 0, st>>>(ws + QB_, 2048ull * 1536, ws + SCAL_ + 0);
    sumsq_kernel<<<dim3(512), 256, 0, st>>>(ws + KB_, 2048ull * 1536, ws + SCAL_ + 1);
    invnorm_kernel<<<dim3(1), 32, 0, st>>>(ws + SCAL_);
    colsum_kernel<<<g1(1536), 256, 0, st>>>(ws + KB_, ws + KSUM_, 2048, 1536);
    for (int h = 0; h < NHEAD; ++h)
      launch_gemm(st, ws + KB_ + h * 384, 1, 1536, ws + VB_ + h * 384, 1536, 1,
                  ws + KVS_ + (size_t)h * 384 * 384, 384, 1,
                  384, 384, 2048, nullptr, nullptr, 1.f, 0);
    for (int h = 0; h < NHEAD; ++h)
      launch_gemm(st, ws + QB_ + h * 384, 1536, 1, ws + KVS_ + (size_t)h * 384 * 384, 384, 1,
                  ws + NUMB_ + h * 384, 1536, 1,
                  2048, 384, 384, nullptr, ws + SCAL_ + 2, 1.f, 0);
    dnm_kernel<<<g1(2048ull * 4), 256, 0, st>>>(ws + QB_, ws + KSUM_, ws + SCAL_ + 2,
                                                ws + DNM_, 2048);
    linattn_update_kernel<<<g1(2048ull * 384), 256, 0, st>>>(ws + XG_, ws + NUMB_, ws + VB_,
        ws + DNM_, bn_g + (size_t)(i + 1) * 384, bn_b + (size_t)(i + 1) * 384, 2048);
  }
  mix_kernel<<<g1(2048ull * 384), 256, 0, st>>>(ws + HMIX_, ws + HS_, ws + XG_,
                                                0.7f, 0.3f, 2048ull * 384);

  // ---- trajectory MHA + summary ----
  run_mha(st, ws + HMIX_, 2048, ws + HMIX_, 2048,
          mha_Wqkv + 1ull * 384 * 1152, mha_bqkv + 1152, mha_Wo + 1ull * 384 * 384, mha_bo + 384,
          ws + QB2_, ws + KB2_, ws + VB2_, ws + OB2_, ws + SCORES_, ws + TRAJ_);
  col_mean_kernel<<<g1(384), 256, 0, st>>>(ws + TRAJ_, 384, 1, 2048, 384, ws + SMALL_ + 0);

  // ---- conv decay path ----
  conv1d_k3_gelu_kernel<<<g1(384ull * 2048), 256, 0, st>>>(ws + HMIX_, 1, 384,
      c1_W, c1_b, ws + CONV1_, 384, 384, 2048);
  conv1d_k3_gelu_kernel<<<g1(192ull * 2048), 256, 0, st>>>(ws + CONV1_, 2048, 1,
      c2_W, c2_b, ws + CONV2_, 384, 192, 2048);
  col_mean_kernel<<<g1(192), 256, 0, st>>>(ws + CONV2_, 1, 2048, 2048, 192, ws + SMALL_ + 384);
  vecmat2_kernel<<<g1(384), 256, 0, st>>>(ws + SMALL_ + 0, 384, ws + SMALL_ + 384, 192,
      op_W, 384, op_b, ws + SMALL_ + 576, 384, 0);

  // ---- ai context MHA, pooled; pressure scores ----
  run_mha(st, ai_h, 2048, user_h, 2048,
          mha_Wqkv, mha_bqkv, mha_Wo, mha_bo,
          ws + QB2_, ws + KB2_, ws + VB2_, ws + OB2_, ws + SCORES_, ws + AICTX_);
  col_mean_kernel<<<g1(384), 256, 0, st>>>(ws + AICTX_, 384, 1, 2048, 384, ws + SMALL_ + 960);
  concat2_rows_kernel<<<g1(2048ull * 768), 256, 0, st>>>(ws + COMB_, ai_h, user_h, 2048, 384, 384);
  launch_gemm(st, ws + COMB_, 768, 1, as1_W, 384, 1, ws + ASBUF_, 384, 1,
              2048, 384, 768, as1_b, nullptr, 1.f, 2);
  gemv_rows_kernel<<<g1(2048), 256, 0, st>>>(ws + ASBUF_, 384, as2_W, 1, as2_b,
                                             out + 2050, 2048, 384, 3);

  // ---- belief context MHA ----
  run_mha(st, ai_h, 2048, belief_h, 1024,
          mha_Wqkv + 2ull * 384 * 1152, mha_bqkv + 2ull * 1152,
          mha_Wo + 2ull * 384 * 384, mha_bo + 2ull * 384,
          ws + QB2_, ws + KB2_, ws + VB2_, ws + OB2_, ws + SCORES_, ws + BCTX_);
  col_mean_kernel<<<g1(384), 256, 0, st>>>(ws + BCTX_, 384, 1, 2048, 384, ws + SMALL_ + 1344);

  // ---- stance delta ----
  vecmat2_kernel<<<g1(384), 256, 0, st>>>(stance_h, 384, stance_h + 2047ull * 384, 384,
      sc1_W, 384, sc1_b, ws + SMALL_ + 1728, 384, 2);
  vecmat2_kernel<<<g1(192), 256, 0, st>>>(ws + SMALL_ + 1728, 384, nullptr, 0,
      sc2_W, 192, sc2_b, ws + SMALL_ + 2112, 192, 0);

  // ---- final head + per-turn scores ----
  final_head_kernel<<<dim3(1), 512, 0, st>>>(ws + SMALL_ + 576, ws + SMALL_ + 960,
      ws + SMALL_ + 1344, ws + SMALL_ + 2112,
      cl1_W, cl1_b, cl_lng, cl_lnb, cl2_W, cl2_b, cl3_W, cl3_b, out);
  gemv_rows_kernel<<<g1(2048), 256, 0, st>>>(ws + AICTX_, 384, tp_W, 1, tp_b,
                                             out + 2, 2048, 384, 3);
}
